// FNO1d_15247133900993
// MI455X (gfx1250) — compile-verified, matrix operand order checked by reading, not run on hardware
//
#include <hip/hip_runtime.h>
#include <hip/hip_bf16.h>
#include <math.h>

typedef __bf16 bf16;
typedef __attribute__((ext_vector_type(16))) __bf16 v16bf;
typedef __attribute__((ext_vector_type(8)))  float  v8f;

#define DEVI __device__ __forceinline__

static constexpr int kB = 2048;
static constexpr int kW = 128;     // channels == spatial length
static constexpr int kModes = 16;
static constexpr int kU = 64;
static constexpr int kLDP = 40;    // LDS pitch for 32-wide K chunks (keeps 16B align, dodges conflicts)

// ---------------- WMMA helpers (CDNA5 wave32) ----------------

DEVI v8f wmma_bf16(v16bf a, v16bf b, v8f c) {
  // D = A(16x32) * B(32x16) + C, fp32 accumulate
  return __builtin_amdgcn_wmma_f32_16x16x32_bf16(false, a, false, b, (short)0, c, false, false);
}

// Load a 16x32 bf16 operand fragment from a row-major LDS tile (pitch kLDP).
// Per ISA 7.12.2: lanes 0-15 -> M=lane, K={0..7,16..23}; lanes 16-31 -> M=lane-16, K={8..15,24..31}.
DEVI v16bf frag_ld(const bf16* p) {
  const int lane = threadIdx.x & 31;
  const bf16* q = p + (lane & 15) * kLDP + ((lane >> 4) << 3);
  v16bf f;
#pragma unroll
  for (int e = 0; e < 8; ++e) f[e] = q[e];
#pragma unroll
  for (int e = 0; e < 8; ++e) f[e + 8] = q[e + 16];
  return f;
}

DEVI v16bf neg_frag(v16bf x) {  // bf16 WMMA has no A/B negate modifier (IU8 only) -> negate in VALU
  v16bf r;
#pragma unroll
  for (int e = 0; e < 16; ++e) r[e] = (bf16)(-(float)x[e]);
  return r;
}

DEVI float gelu_f(float x) { return 0.5f * x * (1.0f + erff(x * 0.707106781186547524f)); }

// ---- CDNA5 async global->LDS tile staging (ASYNCcnt path, bypasses VGPRs) ----
// the LDS destination address comes from the VDST VGPR; low 32 bits of a generic LDS pointer are the
// wave-relative LDS byte address (flat addressing truncation rule).
DEVI void async_cp16(const bf16* __restrict__ g, bf16* l) {
  const unsigned lo = (unsigned)(uintptr_t)l;
  asm volatile("global_load_async_to_lds_b128 %0, %1, off"
               :: "v"(lo), "v"(g) : "memory");
}
DEVI void async_wait0() { asm volatile("s_wait_asynccnt 0x0" ::: "memory"); }

// ---------------- Generic NT GEMM: C = act(A[MxK] * Bt[NxK]^T + bias) ----------------
// TROUT: write transposed (col-major planes) for FFT -> per-mode matrices.

template <int BM, int BN, bool BIAS, bool GELU, bool F32OUT, bool TROUT>
__global__ __launch_bounds__(256) void gemm_nt(const bf16* __restrict__ A,
                                               const bf16* __restrict__ Bt,
                                               const float* __restrict__ bias,
                                               void* __restrict__ Cout,
                                               int M, int N, int K) {
  __shared__ alignas(16) bf16 sA[BM * kLDP];
  __shared__ alignas(16) bf16 sB[BN * kLDP];
  constexpr int GM = (BN >= 128) ? 4 : 8;   // wave grid over M
  constexpr int GN = (BN >= 128) ? 2 : 1;   // wave grid over N
  constexpr int TM = BM / (16 * GM);
  constexpr int TN = BN / (16 * GN);
  const int tid = threadIdx.x;
  const int m0 = blockIdx.y * BM;
  const int n0 = blockIdx.x * BN;
  const int wave = tid >> 5, lane = tid & 31;
  const int wm = wave / GN, wn = wave % GN;

  v8f acc[TM][TN];
#pragma unroll
  for (int i = 0; i < TM; ++i)
#pragma unroll
    for (int j = 0; j < TN; ++j)
#pragma unroll
      for (int e = 0; e < 8; ++e) acc[i][j][e] = 0.0f;

  for (int k0 = 0; k0 < K; k0 += 32) {
    __syncthreads();  // previous iter's frags consumed by WMMA before waves got here
    for (int c = tid; c < BM * 4; c += 256) {
      const int r = c >> 2, cc = (c & 3) << 3;
      async_cp16(&A[(size_t)(m0 + r) * K + (k0 + cc)], &sA[r * kLDP + cc]);
    }
    for (int c = tid; c < BN * 4; c += 256) {
      const int r = c >> 2, cc = (c & 3) << 3;
      async_cp16(&Bt[(size_t)(n0 + r) * K + (k0 + cc)], &sB[r * kLDP + cc]);
    }
    if (k0 + 32 < K) {  // global_prefetch_b8 of next K tile
      __builtin_prefetch(&A[(size_t)(m0 + (tid >> 1)) * K + (k0 + 32)], 0, 3);
      if (tid < BN * 2)
        __builtin_prefetch(&Bt[(size_t)(n0 + (tid >> 1)) * K + (k0 + 32)], 0, 3);
    }
    async_wait0();
    __syncthreads();
    v16bf af[TM], bfv[TN];
#pragma unroll
    for (int i = 0; i < TM; ++i) af[i] = frag_ld(&sA[(wm * TM + i) * 16 * kLDP]);
#pragma unroll
    for (int j = 0; j < TN; ++j) bfv[j] = frag_ld(&sB[(wn * TN + j) * 16 * kLDP]);
#pragma unroll
    for (int i = 0; i < TM; ++i)
#pragma unroll
      for (int j = 0; j < TN; ++j) acc[i][j] = wmma_bf16(af[i], bfv[j], acc[i][j]);
  }

  const int cn = lane & 15;
  const int mh = (lane >> 4) << 3;
#pragma unroll
  for (int i = 0; i < TM; ++i) {
#pragma unroll
    for (int j = 0; j < TN; ++j) {
      const int col = n0 + (wn * TN + j) * 16 + cn;
      const float bv = BIAS ? bias[col] : 0.0f;
#pragma unroll
      for (int e = 0; e < 8; ++e) {
        const int row = m0 + (wm * TM + i) * 16 + mh + e;
        float v = acc[i][j][e] + bv;
        if (GELU) v = gelu_f(v);
        if (F32OUT)
          ((float*)Cout)[(size_t)row * N + col] = v;
        else if (TROUT)
          ((bf16*)Cout)[(size_t)col * M + row] = (bf16)v;   // scatter to mode planes
        else
          ((bf16*)Cout)[(size_t)row * N + col] = (bf16)v;
      }
    }
  }
}

// ---------------- Complex mode mixing: per mode m, Y = X * W (complex), X:(B x C), W^T prepped ----
// planes: 32 planes of (B*W) bf16: [m] real, [16+m] imag, each (b,i) row-major.
// mw layout: [part(r=0,i=1)][m][o][i] bf16.  Output yf: rows (b*128+o), 32 cols [Re m | Im m].

__global__ __launch_bounds__(256) void mix_modes(const bf16* __restrict__ planes,
                                                 const bf16* __restrict__ mw,
                                                 bf16* __restrict__ yf) {
  __shared__ alignas(16) bf16 sXr[128 * kLDP];
  __shared__ alignas(16) bf16 sXi[128 * kLDP];
  __shared__ alignas(16) bf16 sWr[128 * kLDP];
  __shared__ alignas(16) bf16 sWi[128 * kLDP];
  const int mode = blockIdx.x;
  const int b0 = blockIdx.y * 128;
  const size_t PS = (size_t)kB * kW;
  const bf16* Xr = planes + (size_t)mode * PS + (size_t)b0 * kW;
  const bf16* Xi = planes + (size_t)(kModes + mode) * PS + (size_t)b0 * kW;
  const bf16* Wr = mw + (size_t)mode * (kW * kW);
  const bf16* Wi = mw + (size_t)(kModes + mode) * (kW * kW);
  const int tid = threadIdx.x, wave = tid >> 5, lane = tid & 31;
  const int wm = wave >> 1, wn = wave & 1;

  v8f ar[2][4], ai[2][4];
#pragma unroll
  for (int i = 0; i < 2; ++i)
#pragma unroll
    for (int j = 0; j < 4; ++j)
#pragma unroll
      for (int e = 0; e < 8; ++e) { ar[i][j][e] = 0.0f; ai[i][j][e] = 0.0f; }

  for (int k0 = 0; k0 < 128; k0 += 32) {
    __syncthreads();
    for (int c = tid; c < 512; c += 256) {
      const int r = c >> 2, cc = (c & 3) << 3;
      const size_t go = (size_t)r * 128 + k0 + cc;
      const int lo = r * kLDP + cc;
      async_cp16(&Xr[go], &sXr[lo]);
      async_cp16(&Xi[go], &sXi[lo]);
      async_cp16(&Wr[go], &sWr[lo]);
      async_cp16(&Wi[go], &sWi[lo]);
    }
    async_wait0();
    __syncthreads();
    v16bf axr[2], axi[2], axin[2], bwr[4], bwi[4];
#pragma unroll
    for (int i = 0; i < 2; ++i) {
      axr[i] = frag_ld(&sXr[(wm * 2 + i) * 16 * kLDP]);
      axi[i] = frag_ld(&sXi[(wm * 2 + i) * 16 * kLDP]);
      axin[i] = neg_frag(axi[i]);
    }
#pragma unroll
    for (int j = 0; j < 4; ++j) {
      bwr[j] = frag_ld(&sWr[(wn * 4 + j) * 16 * kLDP]);
      bwi[j] = frag_ld(&sWi[(wn * 4 + j) * 16 * kLDP]);
    }
#pragma unroll
    for (int i = 0; i < 2; ++i)
#pragma unroll
      for (int j = 0; j < 4; ++j) {
        ar[i][j] = wmma_bf16(axr[i], bwr[j], ar[i][j]);    // Yr += Xr*Wr
        ar[i][j] = wmma_bf16(axin[i], bwi[j], ar[i][j]);   // Yr -= Xi*Wi
        ai[i][j] = wmma_bf16(axr[i], bwi[j], ai[i][j]);    // Yi += Xr*Wi
        ai[i][j] = wmma_bf16(axi[i], bwr[j], ai[i][j]);    // Yi += Xi*Wr
      }
  }

  const int cn = lane & 15, mh = (lane >> 4) << 3;
#pragma unroll
  for (int i = 0; i < 2; ++i)
#pragma unroll
    for (int j = 0; j < 4; ++j)
#pragma unroll
      for (int e = 0; e < 8; ++e) {
        const int brow = b0 + (wm * 2 + i) * 16 + mh + e;
        const int o = (wn * 4 + j) * 16 + cn;
        const size_t base = ((size_t)brow * kW + o) * 32;
        yf[base + mode] = (bf16)ar[i][j][e];
        yf[base + kModes + mode] = (bf16)ai[i][j][e];
      }
}

// ---------------- Pointwise conv + residual spectral + bias (+gelu), in-place on x ---------------
// Per batch b: out(o,l) = act( W(o,i) @ x_b(i,l) + spec_b(o,l) + bias(o) ).  x transposed via LDS.

template <bool GELU>
__global__ __launch_bounds__(256) void pw_combine(const bf16* __restrict__ Wl,
                                                  bf16* __restrict__ x,
                                                  const bf16* __restrict__ spec,
                                                  const float* __restrict__ bias) {
  __shared__ alignas(16) bf16 sW[128 * kLDP];
  __shared__ alignas(16) bf16 sX[128 * kLDP];   // rows l, cols i-chunk (transposed stage)
  const int b = blockIdx.x;
  bf16* xb = x + (size_t)b * (kW * kW);
  const bf16* specb = spec + (size_t)b * (kW * kW);
  const int tid = threadIdx.x, wave = tid >> 5, lane = tid & 31;
  const int wm = wave >> 1, wn = wave & 1;

  v8f acc[2][4];
#pragma unroll
  for (int i = 0; i < 2; ++i)
#pragma unroll
    for (int j = 0; j < 4; ++j)
#pragma unroll
      for (int e = 0; e < 8; ++e) acc[i][j][e] = 0.0f;

  for (int k0 = 0; k0 < 128; k0 += 32) {
    __syncthreads();
    for (int c = tid; c < 512; c += 256) {  // W tile: straight copy -> async path
      const int r = c >> 2, cc = (c & 3) << 3;
      async_cp16(&Wl[(size_t)r * 128 + k0 + cc], &sW[r * kLDP + cc]);
    }
    {  // stage x chunk (32 i-rows x 128 l) transposed into sX[l][i] (VALU ds-store path)
      const int r = tid >> 3;         // i row 0..31
      const int l0 = (tid & 7) << 4;  // 16 l's per thread
      float4 v0 = *(const float4*)&xb[(size_t)(k0 + r) * 128 + l0];
      float4 v1 = *(const float4*)&xb[(size_t)(k0 + r) * 128 + l0 + 8];
      bf16 tmp[16];
      *(float4*)&tmp[0] = v0;
      *(float4*)&tmp[8] = v1;
#pragma unroll
      for (int t = 0; t < 16; ++t) sX[(l0 + t) * kLDP + r] = tmp[t];
    }
    async_wait0();
    __syncthreads();
    v16bf aw[2], bx[4];
#pragma unroll
    for (int i = 0; i < 2; ++i) aw[i] = frag_ld(&sW[(wm * 2 + i) * 16 * kLDP]);
#pragma unroll
    for (int j = 0; j < 4; ++j) bx[j] = frag_ld(&sX[(wn * 4 + j) * 16 * kLDP]);
#pragma unroll
    for (int i = 0; i < 2; ++i)
#pragma unroll
      for (int j = 0; j < 4; ++j) acc[i][j] = wmma_bf16(aw[i], bx[j], acc[i][j]);
  }

  const int cn = lane & 15, mh = (lane >> 4) << 3;
#pragma unroll
  for (int i = 0; i < 2; ++i)
#pragma unroll
    for (int j = 0; j < 4; ++j)
#pragma unroll
      for (int e = 0; e < 8; ++e) {
        const int o = (wm * 2 + i) * 16 + mh + e;
        const int l = (wn * 4 + j) * 16 + cn;
        const size_t idx = (size_t)o * 128 + l;
        float v = acc[i][j][e] + (float)specb[idx] + bias[o];
        if (GELU) v = gelu_f(v);
        xb[idx] = (bf16)v;
      }
}

// ---------------- Small prep / elementwise kernels ----------------

__global__ void cvt_f2b(const float* __restrict__ in, bf16* __restrict__ out, int n) {
  int i = blockIdx.x * 256 + threadIdx.x;
  if (i < n) out[i] = (bf16)in[i];
}

// sc weights (i,o,m) f32 -> [part][m][o][i] bf16
__global__ void prep_modew(const float* __restrict__ wr, const float* __restrict__ wi,
                           bf16* __restrict__ out) {
  int idx = blockIdx.x * 256 + threadIdx.x;
  if (idx >= kW * kW * kModes) return;
  const int i = idx >> 11;
  const int o = (idx >> 4) & 127;
  const int m = idx & 15;
  const size_t d = ((size_t)m * kW + o) * kW + i;
  out[d] = (bf16)wr[idx];
  out[(size_t)kModes * kW * kW + d] = (bf16)wi[idx];
}

// DFT matrices: Ft (32 x 128): rows 0..15 cos, 16..31 -sin.  Gt (128 x 32): iDFT^T for NT GEMM.
__global__ void prep_dft(bf16* __restrict__ Ft, bf16* __restrict__ Gt) {
  int idx = blockIdx.x * 256 + threadIdx.x;
  const float w0 = 6.283185307179586f / 128.0f;
  if (idx < 4096) {
    const int p = idx >> 7, l = idx & 127, m = p & 15;
    const float a = w0 * (float)(m * l);
    Ft[idx] = (bf16)((p < kModes) ? __cosf(a) : -__sinf(a));
  } else if (idx < 8192) {
    const int j = idx - 4096;
    const int l = j >> 5, p = j & 31, m = p & 15;
    const float a = w0 * (float)(m * l);
    float v;
    if (p == 0) v = 1.0f / 128.0f;                       // DC
    else if (p < kModes) v = (2.0f / 128.0f) * __cosf(a);
    else if (p == kModes) v = 0.0f;                      // imag of DC dropped by irfft
    else v = -(2.0f / 128.0f) * __sinf(a);
    Gt[(size_t)l * 32 + p] = (bf16)v;
  }
}

// Layer-0 mode mix is elementwise (1 input channel)
__global__ void mix0(const bf16* __restrict__ xf0, const float* __restrict__ w0r,
                     const float* __restrict__ w0i, bf16* __restrict__ yf) {
  int idx = blockIdx.x * 256 + threadIdx.x;
  if (idx >= kB * kW * kModes) return;
  const int m = idx & 15;
  const int o = (idx >> 4) & 127;
  const int b = idx >> 11;
  const float xr = (float)xf0[(size_t)b * 32 + m];
  const float xi = (float)xf0[(size_t)b * 32 + kModes + m];
  const float wr = w0r[o * kModes + m];
  const float wi = w0i[o * kModes + m];
  const size_t base = ((size_t)b * kW + o) * 32;
  yf[base + m] = (bf16)(xr * wr - xi * wi);
  yf[base + kModes + m] = (bf16)(xr * wi + xi * wr);
}

// Layer-0 pointwise (1 input channel) + gelu
__global__ void combine0(const bf16* __restrict__ spec, const bf16* __restrict__ x0,
                         const float* __restrict__ w0w, const float* __restrict__ w0b,
                         bf16* __restrict__ x) {
  int idx = blockIdx.x * 256 + threadIdx.x;
  if (idx >= kB * kW * kW) return;
  const int l = idx & 127;
  const int o = (idx >> 7) & 127;
  const int b = idx >> 14;
  float v = (float)spec[idx] + w0w[o] * (float)x0[(size_t)b * kW + l] + w0b[o];
  x[idx] = (bf16)gelu_f(v);
}

// ---------------- Host orchestration ----------------

extern "C" void kernel_launch(void* const* d_in, const int* in_sizes, int n_in,
                              void* d_out, int out_size, void* d_ws, size_t ws_size,
                              hipStream_t stream) {
  (void)in_sizes; (void)n_in; (void)out_size; (void)ws_size;
  const float* X1 = (const float*)d_in[0];
  const float* fc0w = (const float*)d_in[1];
  const float* fc0b = (const float*)d_in[2];
  const float* sc0r = (const float*)d_in[3];
  const float* sc0i = (const float*)d_in[4];
  const float* scr[3] = {(const float*)d_in[5], (const float*)d_in[7], (const float*)d_in[9]};
  const float* sci[3] = {(const float*)d_in[6], (const float*)d_in[8], (const float*)d_in[10]};
  const float* w0w = (const float*)d_in[11];
  const float* w0b = (const float*)d_in[12];
  const float* wpw[3] = {(const float*)d_in[13], (const float*)d_in[15], (const float*)d_in[17]};
  const float* wpb[3] = {(const float*)d_in[14], (const float*)d_in[16], (const float*)d_in[18]};
  const float* fc1w = (const float*)d_in[19];
  const float* fc1b = (const float*)d_in[20];
  const float* fc2w = (const float*)d_in[21];
  const float* fc2b = (const float*)d_in[22];

  char* ws = (char*)d_ws;
  size_t off = 0;
  auto alloc = [&](size_t elems) -> bf16* {
    off = (off + 255) & ~(size_t)255;
    bf16* p = (bf16*)(ws + off);
    off += elems * sizeof(bf16);
    return p;
  };
  bf16* X1b = alloc((size_t)kB * kW);
  bf16* fc0wb = alloc((size_t)kW * kW);
  bf16* fc1wb = alloc((size_t)kW * kW * kW);
  bf16* fc2wb = alloc((size_t)kU * kW);
  bf16* wpwb = alloc((size_t)3 * kW * kW);
  bf16* mwb = alloc((size_t)3 * 2 * kModes * kW * kW);
  bf16* Ft = alloc(32 * 128);
  bf16* Gt = alloc(128 * 32);
  bf16* x0 = alloc((size_t)kB * kW);
  bf16* xf0 = alloc((size_t)kB * 32);
  bf16* h = alloc((size_t)kB * kW);
  bf16* bufX = alloc((size_t)kB * kW * kW);
  bf16* spec = alloc((size_t)kB * kW * kW);
  bf16* planes = alloc((size_t)32 * kB * kW);
  bf16* yf = alloc((size_t)kB * kW * 32);

  auto cvt = [&](const float* src, bf16* dst, int n) {
    cvt_f2b<<<(n + 255) / 256, 256, 0, stream>>>(src, dst, n);
  };
  cvt(X1, X1b, kB * kW);
  cvt(fc0w, fc0wb, kW * kW);
  cvt(fc1w, fc1wb, kW * kW * kW);
  cvt(fc2w, fc2wb, kU * kW);
  for (int l = 0; l < 3; ++l) cvt(wpw[l], wpwb + (size_t)l * kW * kW, kW * kW);
  for (int l = 0; l < 3; ++l)
    prep_modew<<<(kW * kW * kModes + 255) / 256, 256, 0, stream>>>(
        scr[l], sci[l], mwb + (size_t)l * 2 * kModes * kW * kW);
  prep_dft<<<32, 256, 0, stream>>>(Ft, Gt);

  // fc0: x0(b,l) = X1 @ fc0_w^T + b
  gemm_nt<128, 128, true, false, false, false><<<dim3(1, kB / 128), 256, 0, stream>>>(
      X1b, fc0wb, fc0b, x0, kB, kW, kW);

  // layer 0 (1 input channel)
  gemm_nt<128, 32, false, false, false, false><<<dim3(1, kB / 128), 256, 0, stream>>>(
      x0, Ft, nullptr, xf0, kB, 32, kW);
  mix0<<<(kB * kW * kModes + 255) / 256, 256, 0, stream>>>(xf0, sc0r, sc0i, yf);
  gemm_nt<128, 128, false, false, false, false><<<dim3(1, (kB * kW) / 128), 256, 0, stream>>>(
      yf, Gt, nullptr, spec, kB * kW, kW, 32);
  combine0<<<(kB * kW * kW + 255) / 256, 256, 0, stream>>>(spec, x0, w0w, w0b, bufX);

  // layers 1..3
  for (int l = 0; l < 3; ++l) {
    gemm_nt<128, 32, false, false, false, true><<<dim3(1, (kB * kW) / 128), 256, 0, stream>>>(
        bufX, Ft, nullptr, planes, kB * kW, 32, kW);                 // FFT -> mode planes
    mix_modes<<<dim3(kModes, kB / 128), 256, 0, stream>>>(
        planes, mwb + (size_t)l * 2 * kModes * kW * kW, yf);         // complex channel mix
    gemm_nt<128, 128, false, false, false, false><<<dim3(1, (kB * kW) / 128), 256, 0, stream>>>(
        yf, Gt, nullptr, spec, kB * kW, kW, 32);                     // iDFT
    if (l < 2)
      pw_combine<true><<<kB, 256, 0, stream>>>(wpwb + (size_t)l * kW * kW, bufX, spec, wpb[l]);
    else
      pw_combine<false><<<kB, 256, 0, stream>>>(wpwb + (size_t)l * kW * kW, bufX, spec, wpb[l]);
  }

  // fc1 (+gelu), fc2 -> fp32 output
  gemm_nt<128, 128, true, true, false, false><<<dim3(1, kB / 128), 256, 0, stream>>>(
      bufX, fc1wb, fc1b, h, kB, kW, kW * kW);
  gemm_nt<128, 64, true, false, true, false><<<dim3(1, kB / 128), 256, 0, stream>>>(
      h, fc2wb, fc2b, d_out, kB, kU, kW);
}
